// CausalTransformerBlock_35888746726143
// MI455X (gfx1250) — matrix operand order repaired and checked by reading, not compiled, not run on hardware
//
#include <hip/hip_runtime.h>

// ---------------------------------------------------------------------------
// Causal transformer block for MI455X (gfx1250, wave32, WMMA bf16 + TDM).
// B=2, S=2048, D=1024, H=16, HD=64, FF=4096. fp32 in/out, bf16 WMMA compute.
// ---------------------------------------------------------------------------

#define DMODEL 1024
#define SEQ    2048
#define NB     2
#define NH     16
#define HDIM   64
#define FFDIM  4096
#define NTOK   (NB * SEQ)   // 4096 rows

typedef __bf16 bf16;
typedef __attribute__((ext_vector_type(16))) __bf16 v16bf;
typedef __attribute__((ext_vector_type(8)))  __bf16 v8bf;
typedef __attribute__((ext_vector_type(8)))  float  v8f;
typedef __attribute__((ext_vector_type(4)))  unsigned int v4u;
typedef __attribute__((ext_vector_type(8)))  int v8i;
typedef __attribute__((ext_vector_type(4)))  int v4i;

#define NEG_BIG (-3.0e38f)

#if defined(__has_builtin)
#if __has_builtin(__builtin_amdgcn_tensor_load_to_lds) && \
    __has_builtin(__builtin_amdgcn_s_wait_tensorcnt)
#define USE_TDM 1
#endif
#endif
#ifndef USE_TDM
#define USE_TDM 0
#endif

static __device__ __forceinline__ v16bf cat8(v8bf lo, v8bf hi) {
  return __builtin_shufflevector(lo, hi, 0,1,2,3,4,5,6,7,8,9,10,11,12,13,14,15);
}

// A-fragment (16 rows x 32 K, bf16) for v_wmma_f32_16x16x32_bf16.
// Lane (0..15 | 16..31) holds row = lane&15; lane-group g = lane>>4 holds
// K = {g*16 .. g*16+15}. `base` -> (row, kbase).
static __device__ __forceinline__ v16bf load_afrag(const bf16* base, int g) {
  return *(const v16bf*)(base + g * 16);
}

static __device__ __forceinline__ v8f wmma_bf16(v16bf a, v16bf b, v8f c) {
  return __builtin_amdgcn_wmma_f32_16x16x32_bf16(false, a, false, b, (short)0, c,
                                                 false, false);
}

// LDS byte offset of a generic pointer known to point into LDS.
static __device__ __forceinline__ unsigned lds_offset(const void* p) {
  return (unsigned)(unsigned long long)
      (const __attribute__((address_space(3))) char*)p;
}

#if USE_TDM
// Tensor Data Mover: DMA a 2D tile (rows x row_elems of 2-byte elements,
// global row stride = stride_elems) from global memory into LDS.
// Builds D# group0/group1 per CDNA5 ISA 8.3/8.4; groups 2/3/4 unused (2D).
// Toolchain here exposes the 6-arg builtin: (v4u, v8i, v4i, v4i, v8i, cpol).
static __device__ __forceinline__ void tdm_load_2d(const bf16* gsrc,
                                                   unsigned lds_byte,
                                                   unsigned rows,
                                                   unsigned row_elems,
                                                   unsigned long long stride_elems) {
  unsigned long long ga = (unsigned long long)gsrc;
  v4u g0;
  g0[0] = 1u;                                         // count=1, no gather
  g0[1] = lds_byte;                                   // lds_addr
  g0[2] = (unsigned)(ga & 0xffffffffu);               // global_addr[31:0]
  g0[3] = (unsigned)((ga >> 32) & 0x1ffffffu)         // global_addr[56:32]
          | (2u << 30);                               // type=2 (image)
  v8i g1;
  g1[0] = (int)(1u << 16);                            // data_size=1 (2 bytes)
  g1[1] = (int)((row_elems & 0xffffu) << 16);         // tensor_dim0 lo16
  g1[2] = (int)((row_elems >> 16) | ((rows & 0xffffu) << 16));  // td0 hi, td1 lo
  g1[3] = (int)((rows >> 16) | ((row_elems & 0xffffu) << 16));  // td1 hi, tile_dim0
  g1[4] = (int)(rows & 0xffffu);                      // tile_dim1 (tile_dim2=0)
  g1[5] = (int)(unsigned)(stride_elems & 0xffffffffu);          // stride0 lo32
  g1[6] = (int)(unsigned)((stride_elems >> 32) & 0xffffu);      // stride0 hi16
  g1[7] = 0;                                          // stride1 hi (unused)
  v4i g2 = {0, 0, 0, 0};
  v4i g3 = {0, 0, 0, 0};
  v8i g4 = {0, 0, 0, 0, 0, 0, 0, 0};
  __builtin_amdgcn_tensor_load_to_lds(g0, g1, g2, g3, g4, 0);
}
#endif

// ---------------------------------------------------------------------------
// Weight fp32 [K,N]  ->  bf16 transposed [N,K]
// ---------------------------------------------------------------------------
__global__ void cvt_w_transpose(const float* __restrict__ w, bf16* __restrict__ out,
                                int K, int N) {
  long total = (long)K * N;
  for (long i = (long)blockIdx.x * blockDim.x + threadIdx.x; i < total;
       i += (long)gridDim.x * blockDim.x) {
    long n = i / K;
    long k = i - n * K;
    out[i] = (bf16)w[k * N + n];
  }
}

// ---------------------------------------------------------------------------
// LayerNorm over D=1024, fp32 in -> bf16 out. One row per block (256 thr).
// ---------------------------------------------------------------------------
__global__ void __launch_bounds__(256) ln_bf16(const float* __restrict__ x,
                                               const float* __restrict__ gam,
                                               const float* __restrict__ bet,
                                               bf16* __restrict__ out) {
  __shared__ float rs[256];
  __shared__ float rs2[256];
  int row = blockIdx.x, tid = threadIdx.x;
  const float* xr = x + (long)row * DMODEL;
  float s = 0.f, s2 = 0.f;
  for (int i = tid; i < DMODEL; i += 256) {
    float v = xr[i];
    s += v;
    s2 += v * v;
  }
  rs[tid] = s;
  rs2[tid] = s2;
  __syncthreads();
  for (int o = 128; o > 0; o >>= 1) {
    if (tid < o) {
      rs[tid] += rs[tid + o];
      rs2[tid] += rs2[tid + o];
    }
    __syncthreads();
  }
  float mean = rs[0] * (1.f / DMODEL);
  float var = rs2[0] * (1.f / DMODEL) - mean * mean;
  float rstd = rsqrtf(var + 1e-5f);
  bf16* orow = out + (long)row * DMODEL;
  for (int i = tid; i < DMODEL; i += 256)
    orow[i] = (bf16)((xr[i] - mean) * rstd * gam[i] + bet[i]);
}

// ---------------------------------------------------------------------------
// Tiled WMMA GEMM:  out = act(A[M,K] @ Wt[N,K]^T + bias)  (+ residual)
// Block 256 thr (8 waves). Tile: BM=128, BN=128, BK=32. Each wave: 64x32 of C.
// Tiles staged into double-buffered LDS by the Tensor Data Mover (wave 0),
// overlapped with WMMA compute on the other buffer.
// ACT: 0 = none, 1 = exact GELU.  OUTMODE: 0 = bf16 out, 1 = f32 residual out.
// ---------------------------------------------------------------------------
template <int ACT, int OUTMODE>
__global__ void __launch_bounds__(256) gemm_bf16(
    const bf16* __restrict__ A,      // [M,K]
    const bf16* __restrict__ Wt,     // [N,K] (pre-transposed)
    const float* __restrict__ bias,  // [N]
    const float* __restrict__ res,   // [M,N] (OUTMODE==1)
    float* __restrict__ outf,        // [M,N] (OUTMODE==1)
    bf16* __restrict__ outb,         // [M,N] (OUTMODE==0)
    int M, int N, int K) {
  __shared__ __attribute__((aligned(32))) bf16 sA[2][128 * 32];
  __shared__ __attribute__((aligned(32))) bf16 sB[2][128 * 32];

  int tid = threadIdx.x;
  int lane = tid & 31, w = tid >> 5;
  int g = lane >> 4, ln = lane & 15;
  int wm = w >> 2, wn = w & 3;  // 2 x 4 wave grid
  long m0 = (long)blockIdx.y * 128;
  long n0 = (long)blockIdx.x * 128;

  v8f acc[4][2];
  v8f zero = {0.f, 0.f, 0.f, 0.f, 0.f, 0.f, 0.f, 0.f};
#pragma unroll
  for (int mt = 0; mt < 4; ++mt)
#pragma unroll
    for (int nt = 0; nt < 2; ++nt) acc[mt][nt] = zero;

  int nk = K >> 5;

#if USE_TDM
  if (w == 0) {  // TDM issue ignores EXEC; gate per-wave
    tdm_load_2d(A + m0 * K, lds_offset(&sA[0][0]), 128, 32,
                (unsigned long long)K);
    tdm_load_2d(Wt + n0 * K, lds_offset(&sB[0][0]), 128, 32,
                (unsigned long long)K);
    __builtin_amdgcn_s_wait_tensorcnt(0);
  }
  __syncthreads();
#else
  {
    int lr = tid >> 2, lc = (tid & 3) * 8;
#pragma unroll
    for (int pp = 0; pp < 2; ++pp) {
      int r = lr + pp * 64;
      *(v8bf*)(&sA[0][r * 32 + lc]) = *(const v8bf*)(A + (m0 + r) * K + lc);
      *(v8bf*)(&sB[0][r * 32 + lc]) = *(const v8bf*)(Wt + (n0 + r) * K + lc);
    }
    __syncthreads();
  }
#endif

  for (int it = 0; it < nk; ++it) {
    int cur = it & 1, nxt = cur ^ 1;
    int k0n = (it + 1) << 5;
#if USE_TDM
    if (w == 0 && it + 1 < nk) {  // prefetch next K tile via TDM
      tdm_load_2d(A + m0 * K + k0n, lds_offset(&sA[nxt][0]), 128, 32,
                  (unsigned long long)K);
      tdm_load_2d(Wt + n0 * K + k0n, lds_offset(&sB[nxt][0]), 128, 32,
                  (unsigned long long)K);
    }
#else
    if (it + 1 < nk) {
      int lr = tid >> 2, lc = (tid & 3) * 8;
#pragma unroll
      for (int pp = 0; pp < 2; ++pp) {
        int r = lr + pp * 64;
        *(v8bf*)(&sA[nxt][r * 32 + lc]) =
            *(const v8bf*)(A + (m0 + r) * K + k0n + lc);
        *(v8bf*)(&sB[nxt][r * 32 + lc]) =
            *(const v8bf*)(Wt + (n0 + r) * K + k0n + lc);
      }
    }
#endif

    v16bf af[4], bfr[2];
#pragma unroll
    for (int mt = 0; mt < 4; ++mt)
      af[mt] = load_afrag(&sA[cur][(wm * 64 + mt * 16 + ln) * 32], g);
#pragma unroll
    for (int nt = 0; nt < 2; ++nt)
      bfr[nt] = *(const v16bf*)(&sB[cur][(wn * 32 + nt * 16 + ln) * 32 + g * 16]);
#pragma unroll
    for (int mt = 0; mt < 4; ++mt)
#pragma unroll
      for (int nt = 0; nt < 2; ++nt)
        acc[mt][nt] = wmma_bf16(af[mt], bfr[nt], acc[mt][nt]);

    __syncthreads();  // all waves done reading `cur`
#if USE_TDM
    if (w == 0 && it + 1 < nk) __builtin_amdgcn_s_wait_tensorcnt(0);
#endif
    __syncthreads();  // `nxt` buffer now valid for everyone
  }

  // Epilogue. C layout: VGPR slot v <-> row (v + 8*g), column = ln.
#pragma unroll
  for (int mt = 0; mt < 4; ++mt) {
#pragma unroll
    for (int nt = 0; nt < 2; ++nt) {
      long n = n0 + wn * 32 + nt * 16 + ln;
      float bv = bias[n];
#pragma unroll
      for (int vv = 0; vv < 8; ++vv) {
        long m = m0 + wm * 64 + mt * 16 + vv + 8 * g;
        float val = acc[mt][nt][vv] + bv;
        if (ACT == 1) val = 0.5f * val * (1.f + erff(val * 0.7071067811865475f));
        long idx = m * N + n;
        if (OUTMODE == 0)
          outb[idx] = (bf16)val;
        else
          outf[idx] = res[idx] + val;
      }
    }
  }
}

// ---------------------------------------------------------------------------
// Split fused QKV [NTOK, 3*D] bf16 into q,k:[BH, S, 64] and v:[BH, 64, S].
// ---------------------------------------------------------------------------
__global__ void qkv_split(const bf16* __restrict__ qkv, bf16* __restrict__ q,
                          bf16* __restrict__ k, bf16* __restrict__ v) {
  long total = (long)NTOK * 3 * DMODEL;
  for (long i = (long)blockIdx.x * blockDim.x + threadIdx.x; i < total;
       i += (long)gridDim.x * blockDim.x) {
    long t = i / (3 * DMODEL);
    int c = (int)(i - t * (3 * DMODEL));
    int which = c / DMODEL;
    int r = c - which * DMODEL;
    int h = r / HDIM, hd = r - h * HDIM;
    int b = (int)(t / SEQ), s = (int)(t - (long)b * SEQ);
    int bh = b * NH + h;
    bf16 val = qkv[i];
    if (which == 0)
      q[((long)bh * SEQ + s) * HDIM + hd] = val;
    else if (which == 1)
      k[((long)bh * SEQ + s) * HDIM + hd] = val;
    else
      v[((long)bh * HDIM + hd) * SEQ + s] = val;
  }
}

// ---------------------------------------------------------------------------
// Flash attention with causal mask. Block 128 thr = 4 waves, 16 queries/wave,
// key tiles of 32. grid = (B*H, S/64).
// ---------------------------------------------------------------------------
__global__ void __launch_bounds__(128) attn_kernel(const bf16* __restrict__ q,
                                                   const bf16* __restrict__ k,
                                                   const bf16* __restrict__ v,
                                                   bf16* __restrict__ out) {
  __shared__ __attribute__((aligned(32))) bf16 sP[4][16 * 32];

  int tid = threadIdx.x;
  int lane = tid & 31, w = tid >> 5;
  int g = lane >> 4, ln = lane & 15;
  int bh = blockIdx.x;  // 0..B*H-1
  int b = bh / NH, h = bh - b * NH;
  int q0 = blockIdx.y * 64 + w * 16;
  const float scale = 0.125f;  // 1/sqrt(64)

  const bf16* qbase = q + ((long)bh * SEQ + q0 + ln) * HDIM;
  v16bf aq[2];
#pragma unroll
  for (int ks = 0; ks < 2; ++ks) aq[ks] = load_afrag(qbase + ks * 32, g);

  v8f zero = {0.f, 0.f, 0.f, 0.f, 0.f, 0.f, 0.f, 0.f};
  v8f o[4];
#pragma unroll
  for (int t = 0; t < 4; ++t) o[t] = zero;
  float mrow[8], lrow[8];
#pragma unroll
  for (int vv = 0; vv < 8; ++vv) {
    mrow[vv] = NEG_BIG;
    lrow[vv] = 0.f;
  }

  int ktmax = (q0 + 15) >> 5;
  for (int kt = 0; kt <= ktmax; ++kt) {
    int kb = kt * 32;
    v8f s0 = zero, s1 = zero;
#pragma unroll
    for (int ks = 0; ks < 2; ++ks) {
      const bf16* kr0 = k + ((long)bh * SEQ + kb + ln) * HDIM + ks * 32;
      v16bf bk0 = *(const v16bf*)(kr0 + g * 16);
      s0 = wmma_bf16(aq[ks], bk0, s0);
      const bf16* kr1 = k + ((long)bh * SEQ + kb + 16 + ln) * HDIM + ks * 32;
      v16bf bk1 = *(const v16bf*)(kr1 + g * 16);
      s1 = wmma_bf16(aq[ks], bk1, s1);
    }
    float tmax[8];
#pragma unroll
    for (int vv = 0; vv < 8; ++vv) {
      int qrow = q0 + vv + 8 * g;
      float a0 = s0[vv] * scale;
      float a1 = s1[vv] * scale;
      if (kb + ln > qrow) a0 = NEG_BIG;
      if (kb + 16 + ln > qrow) a1 = NEG_BIG;
      s0[vv] = a0;
      s1[vv] = a1;
      tmax[vv] = fmaxf(a0, a1);
    }
#pragma unroll
    for (int m = 1; m < 16; m <<= 1)
#pragma unroll
      for (int vv = 0; vv < 8; ++vv)
        tmax[vv] = fmaxf(tmax[vv], __shfl_xor(tmax[vv], m, 32));

    float rs[8];
#pragma unroll
    for (int vv = 0; vv < 8; ++vv) {
      float mn = fmaxf(mrow[vv], tmax[vv]);
      float alpha = __expf(mrow[vv] - mn);
      float p0 = __expf(s0[vv] - mn);
      float p1 = __expf(s1[vv] - mn);
      mrow[vv] = mn;
      int prow = vv + 8 * g;
      sP[w][prow * 32 + ln] = (bf16)p0;  // C layout -> LDS (row, col)
      sP[w][prow * 32 + 16 + ln] = (bf16)p1;
      rs[vv] = p0 + p1;
#pragma unroll
      for (int t = 0; t < 4; ++t) o[t][vv] *= alpha;
      lrow[vv] *= alpha;
    }
#pragma unroll
    for (int m = 1; m < 16; m <<= 1)
#pragma unroll
      for (int vv = 0; vv < 8; ++vv) rs[vv] += __shfl_xor(rs[vv], m, 32);
#pragma unroll
    for (int vv = 0; vv < 8; ++vv) lrow[vv] += rs[vv];

    // wave-private LDS transpose: wait for our DS stores before A-frag reads
    asm volatile("s_wait_dscnt 0" ::: "memory");
    v16bf ap = load_afrag(&sP[w][ln * 32], g);
#pragma unroll
    for (int t = 0; t < 4; ++t) {
      const bf16* vr = v + ((long)bh * HDIM + t * 16 + ln) * SEQ + kb;
      v16bf bv = *(const v16bf*)(vr + g * 16);
      o[t] = wmma_bf16(ap, bv, o[t]);
    }
    asm volatile("s_wait_dscnt 0" ::: "memory");
  }

  float inv[8];
#pragma unroll
  for (int vv = 0; vv < 8; ++vv) inv[vv] = 1.f / lrow[vv];
#pragma unroll
  for (int t = 0; t < 4; ++t) {
#pragma unroll
    for (int vv = 0; vv < 8; ++vv) {
      int qrow = q0 + vv + 8 * g;
      int d = h * HDIM + t * 16 + ln;
      out[((long)b * SEQ + qrow) * DMODEL + d] = (bf16)(o[t][vv] * inv[vv]);
    }
  }
}

// ---------------------------------------------------------------------------
// Host orchestration
// ---------------------------------------------------------------------------
extern "C" void kernel_launch(void* const* d_in, const int* in_sizes, int n_in,
                              void* d_out, int out_size, void* d_ws, size_t ws_size,
                              hipStream_t stream) {
  (void)in_sizes; (void)n_in; (void)out_size; (void)ws_size;
  const float* x     = (const float*)d_in[0];
  const float* ln1g  = (const float*)d_in[1];
  const float* ln1b  = (const float*)d_in[2];
  const float* wqkv  = (const float*)d_in[3];
  const float* bqkv  = (const float*)d_in[4];
  const float* wproj = (const float*)d_in[5];
  const float* bproj = (const float*)d_in[6];
  const float* ln2g  = (const float*)d_in[7];
  const float* ln2b  = (const float*)d_in[8];
  const float* wfc1  = (const float*)d_in[9];
  const float* bfc1  = (const float*)d_in[10];
  const float* wfc2  = (const float*)d_in[11];
  const float* bfc2  = (const float*)d_in[12];
  float* out = (float*)d_out;

  char* ws = (char*)d_ws;
  size_t off = 0;
  auto alloc = [&](size_t bytes) -> char* {
    char* p = ws + off;
    off += (bytes + 255) & ~((size_t)255);
    return p;
  };
  bf16* wqkvT = (bf16*)alloc((size_t)3 * DMODEL * DMODEL * 2);
  bf16* wprojT = (bf16*)alloc((size_t)DMODEL * DMODEL * 2);
  bf16* wfc1T = (bf16*)alloc((size_t)FFDIM * DMODEL * 2);
  bf16* wfc2T = (bf16*)alloc((size_t)DMODEL * FFDIM * 2);
  bf16* h1 = (bf16*)alloc((size_t)NTOK * DMODEL * 2);
  bf16* qkvb = (bf16*)alloc((size_t)NTOK * 3 * DMODEL * 2);
  bf16* qb = (bf16*)alloc((size_t)NTOK * DMODEL * 2);
  bf16* kb = (bf16*)alloc((size_t)NTOK * DMODEL * 2);
  bf16* vb = (bf16*)alloc((size_t)NTOK * DMODEL * 2);
  bf16* attnb = (bf16*)alloc((size_t)NTOK * DMODEL * 2);
  float* x1 = (float*)alloc((size_t)NTOK * DMODEL * 4);
  bf16* h2 = (bf16*)alloc((size_t)NTOK * DMODEL * 2);
  bf16* gbuf = (bf16*)alloc((size_t)NTOK * FFDIM * 2);

  cvt_w_transpose<<<2048, 256, 0, stream>>>(wqkv, wqkvT, DMODEL, 3 * DMODEL);
  cvt_w_transpose<<<2048, 256, 0, stream>>>(wproj, wprojT, DMODEL, DMODEL);
  cvt_w_transpose<<<2048, 256, 0, stream>>>(wfc1, wfc1T, DMODEL, FFDIM);
  cvt_w_transpose<<<2048, 256, 0, stream>>>(wfc2, wfc2T, FFDIM, DMODEL);

  ln_bf16<<<NTOK, 256, 0, stream>>>(x, ln1g, ln1b, h1);

  gemm_bf16<0, 0><<<dim3(3 * DMODEL / 128, NTOK / 128), 256, 0, stream>>>(
      h1, wqkvT, bqkv, nullptr, nullptr, qkvb, NTOK, 3 * DMODEL, DMODEL);

  qkv_split<<<2048, 256, 0, stream>>>(qkvb, qb, kb, vb);

  attn_kernel<<<dim3(NB * NH, SEQ / 64), 128, 0, stream>>>(qb, kb, vb, attnb);

  gemm_bf16<0, 1><<<dim3(DMODEL / 128, NTOK / 128), 256, 0, stream>>>(
      attnb, wprojT, bproj, x, x1, nullptr, NTOK, DMODEL, DMODEL);

  ln_bf16<<<NTOK, 256, 0, stream>>>(x1, ln2g, ln2b, h2);

  gemm_bf16<1, 0><<<dim3(FFDIM / 128, NTOK / 128), 256, 0, stream>>>(
      h2, wfc1T, bfc1, nullptr, nullptr, gbuf, NTOK, FFDIM, DMODEL);

  gemm_bf16<0, 1><<<dim3(DMODEL / 128, NTOK / 128), 256, 0, stream>>>(
      gbuf, wfc2T, bfc2, x1, out, nullptr, NTOK, DMODEL, FFDIM);
}
